// InternImageLayer_4028679324217
// MI455X (gfx1250) — compile-verified, matrix operand order checked by reading, not run on hardware
//
#include <hip/hip_runtime.h>
#include <hip/hip_bf16.h>
#include <math.h>

// ---------------- problem constants ----------------
#define N_DIM 4
#define H_DIM 56
#define W_DIM 56
#define C_DIM 256
#define G_DIM 16
#define GC_DIM 16
#define P_DIM 9
#define HID_DIM 1024
#define T_DIM (N_DIM * H_DIM * W_DIM)   // 12544 tokens
#define EPSF 1e-6f

typedef unsigned short ushort;
typedef unsigned int uint;
typedef __attribute__((ext_vector_type(16))) __bf16  v16bf;
typedef __attribute__((ext_vector_type(8)))  float   v8f;
typedef __attribute__((ext_vector_type(8)))  ushort  ushort8;

// ---------------- helpers ----------------
static __device__ __forceinline__ ushort f2bf(float f) {
  uint u = __float_as_uint(f);
  u += 0x7FFFu + ((u >> 16) & 1u);           // round-to-nearest-even
  return (ushort)(u >> 16);
}
static __device__ __forceinline__ float bf2f(ushort b) {
  return __uint_as_float(((uint)b) << 16);
}
static __device__ __forceinline__ float gelu_exact(float x) {
  return 0.5f * x * (1.0f + erff(x * 0.70710678118654752f));
}

struct BFrag { ushort8 lo, hi; };
// Load a 16-element bf16 fragment as two contiguous 16B chunks
// (CDNA5 16-bit WMMA operand striping: per-lane K chunks {0..7,16..23}
// for lanes 0-15 and {8..15,24..31} for lanes 16-31).
static __device__ __forceinline__ v16bf load_frag(const ushort* p0, const ushort* p1) {
  BFrag f;
  f.lo = *(const ushort8*)p0;
  f.hi = *(const ushort8*)p1;
  return __builtin_bit_cast(v16bf, f);
}

// ---------------- weight prep: fp32 [Cin,Cout] -> bf16 [Cout,Cin] ----------------
__global__ void wprep_kernel(const float* __restrict__ W, ushort* __restrict__ Wt,
                             int Cin, int Cout) {
  int i = blockIdx.x * blockDim.x + threadIdx.x;
  if (i >= Cin * Cout) return;
  int n = i / Cin, k = i % Cin;
  Wt[i] = f2bf(W[(size_t)k * Cout + n]);
}

// ---------------- LayerNorm (f32 in, bf16 out), one wave32 per token ----------------
__global__ void ln_bf16_kernel(const float* __restrict__ x,
                               const float* __restrict__ g,
                               const float* __restrict__ b,
                               ushort* __restrict__ y) {
  int wave = (blockIdx.x * blockDim.x + threadIdx.x) >> 5;
  int lane = threadIdx.x & 31;
  if (wave >= T_DIM) return;
  const float* row = x + (size_t)wave * C_DIM;
  float v[8];
  float s = 0.f, ss = 0.f;
#pragma unroll
  for (int i = 0; i < 8; ++i) {
    v[i] = row[lane * 8 + i];
    s += v[i]; ss += v[i] * v[i];
  }
#pragma unroll
  for (int off = 16; off; off >>= 1) {
    s  += __shfl_xor(s,  off, 32);
    ss += __shfl_xor(ss, off, 32);
  }
  float mu   = s * (1.0f / C_DIM);
  float var  = ss * (1.0f / C_DIM) - mu * mu;
  float rstd = rsqrtf(var + EPSF);
  ushort* outp = y + (size_t)wave * C_DIM;
#pragma unroll
  for (int i = 0; i < 8; ++i) {
    int c = lane * 8 + i;
    outp[c] = f2bf((v[i] - mu) * rstd * g[c] + b[c]);
  }
}

// ---------------- WMMA bf16 GEMM: out[T,Cout] = A[T,Cin] @ Wt[Cout,Cin]^T + bias ---------
// One wave32 per 16x(16*NT) output tile: the A fragment is loaded once per
// 32-deep k-step and reused against NT B fragments (NT wmma per A load pair),
// amortizing activation traffic. 8 waves/block -> 128 rows per block.
#define EPI_STORE_F32 0
#define EPI_GELU_BF16 1
#define EPI_RESID_F32 2

template <int NT>
__global__ __launch_bounds__(256)
void gemm_wmma_bf16(const ushort* __restrict__ A,
                    const ushort* __restrict__ Wt,
                    const float* __restrict__ bias,
                    const float* __restrict__ resid,
                    float* __restrict__ outF,
                    ushort* __restrict__ outB,
                    int Cin, int Cout, int mode) {
  int lane   = threadIdx.x & 31;
  int waveId = threadIdx.x >> 5;
  int m0 = blockIdx.y * 128 + waveId * 16;
  int n0 = blockIdx.x * (16 * NT);
  int lane16 = lane & 15;
  int sel = (lane < 16) ? 0 : 8;

  const ushort* aRow = A + (size_t)(m0 + lane16) * Cin;
  const ushort* bRow[NT];
#pragma unroll
  for (int j = 0; j < NT; ++j)
    bRow[j] = Wt + (size_t)(n0 + j * 16 + lane16) * Cin;

  v8f acc[NT];
#pragma unroll
  for (int j = 0; j < NT; ++j) acc[j] = (v8f){};

  for (int k0 = 0; k0 < Cin; k0 += 32) {
    __builtin_prefetch(aRow + k0 + 64, 0, 0);   // global_prefetch_b8 for next A tile
    v16bf a = load_frag(aRow + k0 + sel, aRow + k0 + 16 + sel);
#pragma unroll
    for (int j = 0; j < NT; ++j) {
      v16bf b = load_frag(bRow[j] + k0 + sel, bRow[j] + k0 + 16 + sel);
      acc[j] = __builtin_amdgcn_wmma_f32_16x16x32_bf16(
          /*neg_a=*/false, a, /*neg_b=*/false, b,
          /*c_mod=*/(short)0, acc[j], /*reuse_a=*/false, /*reuse_b=*/false);
    }
  }

  // C/D layout: VGPR r, lanes 0-15 -> row m0+r, lanes 16-31 -> row m0+8+r
  int rbase = m0 + ((lane < 16) ? 0 : 8);
#pragma unroll
  for (int j = 0; j < NT; ++j) {
    int col = n0 + j * 16 + lane16;
    float bcol = bias[col];
#pragma unroll
    for (int r = 0; r < 8; ++r) {
      int row = rbase + r;
      float v = acc[j][r] + bcol;
      size_t idx = (size_t)row * Cout + col;
      if (mode == EPI_GELU_BF16) {
        outB[idx] = f2bf(gelu_exact(v));
      } else if (mode == EPI_RESID_F32) {
        outF[idx] = v + resid[idx];
      } else {
        outF[idx] = v;
      }
    }
  }
}

// ---------------- depthwise 3x3 conv + LN + GELU (offset/mask branch) ----------------
// One block per token, one thread per channel.
__global__ __launch_bounds__(256)
void dw_ln_gelu_kernel(const ushort* __restrict__ ln1x,
                       const float* __restrict__ dw_w,
                       const float* __restrict__ dw_b,
                       const float* __restrict__ dwln_g,
                       const float* __restrict__ dwln_b,
                       ushort* __restrict__ x1) {
  int token = blockIdx.x;
  int c = threadIdx.x;
  int n  = token / (H_DIM * W_DIM);
  int hw = token % (H_DIM * W_DIM);
  int h = hw / W_DIM, w = hw % W_DIM;

  float acc = dw_b[c];
#pragma unroll
  for (int kh = 0; kh < 3; ++kh) {
    int y = h + kh - 1;
    if ((unsigned)y >= (unsigned)H_DIM) continue;
#pragma unroll
    for (int kw = 0; kw < 3; ++kw) {
      int x = w + kw - 1;
      if ((unsigned)x >= (unsigned)W_DIM) continue;
      size_t src = ((size_t)(n * H_DIM + y) * W_DIM + x) * C_DIM + c;
      acc += bf2f(ln1x[src]) * dw_w[(kh * 3 + kw) * C_DIM + c];
    }
  }

  __shared__ float s_sum[8], s_sq[8];
  float s = acc, ss = acc * acc;
#pragma unroll
  for (int off = 16; off; off >>= 1) {
    s  += __shfl_xor(s,  off, 32);
    ss += __shfl_xor(ss, off, 32);
  }
  if ((threadIdx.x & 31) == 0) { s_sum[threadIdx.x >> 5] = s; s_sq[threadIdx.x >> 5] = ss; }
  __syncthreads();
  float ts = 0.f, tss = 0.f;
#pragma unroll
  for (int i = 0; i < 8; ++i) { ts += s_sum[i]; tss += s_sq[i]; }
  float mu   = ts * (1.0f / C_DIM);
  float var  = tss * (1.0f / C_DIM) - mu * mu;
  float rstd = rsqrtf(var + EPSF);
  float yv = (acc - mu) * rstd * dwln_g[c] + dwln_b[c];
  x1[(size_t)token * C_DIM + c] = f2bf(gelu_exact(yv));
}

// ---------------- softmax over P=9 per (token, group), in place ----------------
__global__ void softmax_p_kernel(float* __restrict__ m) {
  int t = blockIdx.x * blockDim.x + threadIdx.x;
  if (t >= T_DIM * G_DIM) return;
  float* p = m + (size_t)t * P_DIM;
  float e[P_DIM];
  float mx = p[0];
#pragma unroll
  for (int i = 1; i < P_DIM; ++i) mx = fmaxf(mx, p[i]);
  float s = 0.f;
#pragma unroll
  for (int i = 0; i < P_DIM; ++i) { e[i] = __expf(p[i] - mx); s += e[i]; }
  float inv = 1.0f / s;
#pragma unroll
  for (int i = 0; i < P_DIM; ++i) p[i] = e[i] * inv;
}

// ---------------- deformable bilinear sampling + mask combine ----------------
// One thread per (token, group); accumulates 16 channels; writes bf16 for outproj GEMM.
__global__ void dcn_sample_kernel(const float* __restrict__ xproj,
                                  const float* __restrict__ offset,
                                  const float* __restrict__ mask,
                                  ushort* __restrict__ dcnout) {
  int t = blockIdx.x * blockDim.x + threadIdx.x;
  if (t >= T_DIM * G_DIM) return;
  int token = t / G_DIM, g = t % G_DIM;
  int n  = token / (H_DIM * W_DIM);
  int hw = token % (H_DIM * W_DIM);
  int h = hw / W_DIM, w = hw % W_DIM;

  const float* off = offset + (size_t)t * P_DIM * 2;
  const float* msk = mask   + (size_t)t * P_DIM;

  float acc[GC_DIM];
#pragma unroll
  for (int c = 0; c < GC_DIM; ++c) acc[c] = 0.f;

#pragma unroll
  for (int p = 0; p < P_DIM; ++p) {
    // padded-coord pixel center: px = w + 1 + (p/3 - 1) + off_x  (Win=58 padded)
    float px = (float)(w + (p / 3)) + off[2 * p + 0];
    float py = (float)(h + (p % 3)) + off[2 * p + 1];
    float x0f = floorf(px), y0f = floorf(py);
    float lx = px - x0f, ly = py - y0f;
    int x0 = (int)x0f, y0 = (int)y0f;
    float m = msk[p];
    float wgt[4] = { (1.f - ly) * (1.f - lx), (1.f - ly) * lx,
                     ly * (1.f - lx),         ly * lx };
    int ys[2] = { y0, y0 + 1 };
    int xs[2] = { x0, x0 + 1 };
#pragma unroll
    for (int cy = 0; cy < 2; ++cy) {
#pragma unroll
      for (int cx = 0; cx < 2; ++cx) {
        int yi = ys[cy], xi = xs[cx];
        // interior of the zero-padded 58x58 image: padded idx in [1,56]
        if (yi < 1 || yi > H_DIM || xi < 1 || xi > W_DIM) continue;
        float wm = m * wgt[cy * 2 + cx];
        const float* src = xproj +
            (((size_t)(n * H_DIM + (yi - 1)) * W_DIM + (xi - 1)) * C_DIM + g * GC_DIM);
#pragma unroll
        for (int c = 0; c < GC_DIM; ++c) acc[c] += wm * src[c];
      }
    }
  }
  ushort* outp = dcnout + (size_t)token * C_DIM + g * GC_DIM;
#pragma unroll
  for (int c = 0; c < GC_DIM; ++c) outp[c] = f2bf(acc[c]);
}

// ---------------- host-side launch ----------------
extern "C" void kernel_launch(void* const* d_in, const int* in_sizes, int n_in,
                              void* d_out, int out_size, void* d_ws, size_t ws_size,
                              hipStream_t stream) {
  const float* x_in      = (const float*)d_in[0];
  const float* ln1_g     = (const float*)d_in[1];
  const float* ln1_b     = (const float*)d_in[2];
  const float* dw_w      = (const float*)d_in[3];
  const float* dw_b      = (const float*)d_in[4];
  const float* dwln_g    = (const float*)d_in[5];
  const float* dwln_b    = (const float*)d_in[6];
  const float* off_w     = (const float*)d_in[7];
  const float* off_b     = (const float*)d_in[8];
  const float* mask_w    = (const float*)d_in[9];
  const float* mask_b    = (const float*)d_in[10];
  const float* inproj_w  = (const float*)d_in[11];
  const float* inproj_b  = (const float*)d_in[12];
  const float* outproj_w = (const float*)d_in[13];
  const float* outproj_b = (const float*)d_in[14];
  const float* ln2_g     = (const float*)d_in[15];
  const float* ln2_b     = (const float*)d_in[16];
  const float* fc1_w     = (const float*)d_in[17];
  const float* fc1_b     = (const float*)d_in[18];
  const float* fc2_w     = (const float*)d_in[19];
  const float* fc2_b     = (const float*)d_in[20];
  float* out = (float*)d_out;

  // workspace carve-up (256B aligned)
  char* ws = (char*)d_ws;
  size_t cur = 0;
  auto alloc = [&](size_t bytes) -> void* {
    void* p = ws + cur;
    cur = (cur + bytes + 255) & ~((size_t)255);
    return p;
  };
  ushort* ln1x    = (ushort*)alloc((size_t)T_DIM * C_DIM * 2);
  float*  xproj   = (float*) alloc((size_t)T_DIM * C_DIM * 4);
  ushort* x1      = (ushort*)alloc((size_t)T_DIM * C_DIM * 2);
  float*  offbuf  = (float*) alloc((size_t)T_DIM * (G_DIM * P_DIM * 2) * 4);
  float*  maskbuf = (float*) alloc((size_t)T_DIM * (G_DIM * P_DIM) * 4);
  ushort* dcnout  = (ushort*)alloc((size_t)T_DIM * C_DIM * 2);
  float*  xres    = (float*) alloc((size_t)T_DIM * C_DIM * 4);
  ushort* ln2x    = (ushort*)alloc((size_t)T_DIM * C_DIM * 2);
  ushort* h1      = (ushort*)alloc((size_t)T_DIM * HID_DIM * 2);
  ushort* inprojT = (ushort*)alloc((size_t)C_DIM * C_DIM * 2);
  ushort* offT    = (ushort*)alloc((size_t)(G_DIM * P_DIM * 2) * C_DIM * 2);
  ushort* maskT   = (ushort*)alloc((size_t)(G_DIM * P_DIM) * C_DIM * 2);
  ushort* outprojT= (ushort*)alloc((size_t)C_DIM * C_DIM * 2);
  ushort* fc1T    = (ushort*)alloc((size_t)HID_DIM * C_DIM * 2);
  ushort* fc2T    = (ushort*)alloc((size_t)C_DIM * HID_DIM * 2);

  const int OFF_C = G_DIM * P_DIM * 2;  // 288
  const int MSK_C = G_DIM * P_DIM;      // 144

  // 1) weight prep (bf16 + transpose to [Cout,Cin])
  auto wprep = [&](const float* W, ushort* Wt, int Cin, int Cout) {
    int total = Cin * Cout;
    wprep_kernel<<<(total + 255) / 256, 256, 0, stream>>>(W, Wt, Cin, Cout);
  };
  wprep(inproj_w,  inprojT,  C_DIM, C_DIM);
  wprep(off_w,     offT,     C_DIM, OFF_C);
  wprep(mask_w,    maskT,    C_DIM, MSK_C);
  wprep(outproj_w, outprojT, C_DIM, C_DIM);
  wprep(fc1_w,     fc1T,     C_DIM, HID_DIM);
  wprep(fc2_w,     fc2T,     HID_DIM, C_DIM);

  const int lnBlocks = (T_DIM * 32) / 256;  // one wave32 per token

  // 2) ln1(x) -> bf16
  ln_bf16_kernel<<<lnBlocks, 256, 0, stream>>>(x_in, ln1_g, ln1_b, ln1x);

  // 3) inproj GEMM: xproj = ln1x @ inproj + b (f32)   (Cout=256, NT=4)
  gemm_wmma_bf16<4><<<dim3(C_DIM / 64, T_DIM / 128), 256, 0, stream>>>(
      ln1x, inprojT, inproj_b, nullptr, xproj, nullptr, C_DIM, C_DIM, EPI_STORE_F32);

  // 4) depthwise conv + LN + GELU -> x1 bf16
  dw_ln_gelu_kernel<<<T_DIM, 256, 0, stream>>>(ln1x, dw_w, dw_b, dwln_g, dwln_b, x1);

  // 5) offset GEMM (Cout=288 = 9*32, NT=2, f32)
  gemm_wmma_bf16<2><<<dim3(OFF_C / 32, T_DIM / 128), 256, 0, stream>>>(
      x1, offT, off_b, nullptr, offbuf, nullptr, C_DIM, OFF_C, EPI_STORE_F32);

  // 6) mask logits GEMM (Cout=144 = 9*16, NT=1, f32) then softmax over P
  gemm_wmma_bf16<1><<<dim3(MSK_C / 16, T_DIM / 128), 256, 0, stream>>>(
      x1, maskT, mask_b, nullptr, maskbuf, nullptr, C_DIM, MSK_C, EPI_STORE_F32);
  softmax_p_kernel<<<(T_DIM * G_DIM + 255) / 256, 256, 0, stream>>>(maskbuf);

  // 7) deformable bilinear sampling + mask combine -> bf16
  dcn_sample_kernel<<<(T_DIM * G_DIM + 255) / 256, 256, 0, stream>>>(
      xproj, offbuf, maskbuf, dcnout);

  // 8) outproj GEMM + residual(x) -> xres f32   (Cout=256, NT=4)
  gemm_wmma_bf16<4><<<dim3(C_DIM / 64, T_DIM / 128), 256, 0, stream>>>(
      dcnout, outprojT, outproj_b, x_in, xres, nullptr, C_DIM, C_DIM, EPI_RESID_F32);

  // 9) ln2(xres) -> bf16
  ln_bf16_kernel<<<lnBlocks, 256, 0, stream>>>(xres, ln2_g, ln2_b, ln2x);

  // 10) fc1 GEMM + GELU -> h1 bf16 (Cout=1024, NT=4)
  gemm_wmma_bf16<4><<<dim3(HID_DIM / 64, T_DIM / 128), 256, 0, stream>>>(
      ln2x, fc1T, fc1_b, nullptr, nullptr, h1, C_DIM, HID_DIM, EPI_GELU_BF16);

  // 11) fc2 GEMM + residual(xres) -> d_out f32 (Cin=1024, Cout=256, NT=4)
  gemm_wmma_bf16<4><<<dim3(C_DIM / 64, T_DIM / 128), 256, 0, stream>>>(
      h1, fc2T, fc2_b, xres, out, nullptr, HID_DIM, C_DIM, EPI_RESID_F32);

  (void)in_sizes; (void)n_in; (void)out_size; (void)ws_size;
}
